// MultiheadAttention_48988396978714
// MI455X (gfx1250) — matrix operand order repaired and checked by reading, not compiled, not run on hardware
//
#include <hip/hip_runtime.h>

#define BS     4
#define SEQ    2048
#define DM     1024
#define HEADS  16
#define DK     64
#define ROWS   (BS * SEQ)

typedef __attribute__((ext_vector_type(16))) __bf16         v16bf;
typedef __attribute__((ext_vector_type(8)))  float          v8f;
typedef __attribute__((ext_vector_type(8)))  unsigned short v8us;
typedef __attribute__((ext_vector_type(16))) unsigned short v16us;

#if __has_builtin(__builtin_amdgcn_global_load_async_to_lds_b128) && \
    __has_builtin(__builtin_amdgcn_s_wait_asynccnt)
#define USE_ASYNC_LDS 1
#endif

// builtin signature (from clang diagnostic): arg0 = vector-of-4-int ptr in
// addrspace(1), arg1 = same in addrspace(3), then imm offset, imm cpol.
typedef __attribute__((__vector_size__(4 * sizeof(int)))) int vi4;
typedef __attribute__((address_space(1))) vi4 gbl_vi4;
typedef __attribute__((address_space(3))) vi4 lds_vi4;
static __device__ __forceinline__ lds_vi4* to_lds(void* p) {
    return (lds_vi4*)(unsigned int)(unsigned long long)p;
}
static __device__ __forceinline__ gbl_vi4* to_gbl(const void* p) {
    return (gbl_vi4*)(unsigned long long)p;
}

static __device__ __forceinline__ v16bf as_bf16(v16us u) {
    return __builtin_bit_cast(v16bf, u);
}
static __device__ __forceinline__ unsigned short f2bf(float f) {
    unsigned int u = __builtin_bit_cast(unsigned int, f);
    u += 0x7FFFu + ((u >> 16) & 1u);   // round-to-nearest-even
    return (unsigned short)(u >> 16);
}
static __device__ __forceinline__ v16us cat8(v8us lo, v8us hi) {
    v16us u;
#pragma unroll
    for (int i = 0; i < 8; ++i) { u[i] = lo[i]; u[i + 8] = hi[i]; }
    return u;
}
static __device__ __forceinline__ v8f wmma_bf16(v16bf a, v16bf b, v8f c) {
    // D = A(16x32 bf16) * B(32x16 bf16) + C(16x16 f32)
    return __builtin_amdgcn_wmma_f32_16x16x32_bf16(
        false, a, false, b, (short)0, c, false, false);
}

// ---------------------------------------------------------------------------
// fp32 -> bf16 conversion
// ---------------------------------------------------------------------------
__global__ __launch_bounds__(256) void cvt_bf16_kernel(
    const float* __restrict__ src, unsigned short* __restrict__ dst, int n) {
    int i = blockIdx.x * 256 + threadIdx.x;
    if (i < n) dst[i] = f2bf(src[i]);
}

// ---------------------------------------------------------------------------
// GEMM: C[M,N] = A[M,K](bf16) * B[K,N](bf16) + bias[N]; fp32 output.
// Block = 256 threads (8 waves) -> 128x128 output tile.
// Wave (mw, nw) computes 32x64 (8 WMMA accumulators).
// K-loop in steps of 32, double-buffered LDS staging.
// A/B tiles staged with GLOBAL_LOAD_ASYNC_TO_LDS_B128 when available
// (ASYNCcnt + barrier), else synchronous copies.
// LDS strides padded (40 / 136 halfwords) for bank spread, 16B aligned.
// ---------------------------------------------------------------------------
#define GBM 128
#define GBN 128
#define GBK 32
#define ALD (GBK + 8)    // 40
#define BLD (GBN + 8)    // 136

__global__ __launch_bounds__(256) void gemm_bf16_kernel(
    const unsigned short* __restrict__ A, const unsigned short* __restrict__ B,
    const float* __restrict__ bias, float* __restrict__ C,
    int M, int N, int K) {
    const int tid  = threadIdx.x;
    const int lane = tid & 31;
    const int wave = tid >> 5;
    const int half = lane >> 4;
    const int l15  = lane & 15;

    const int mw = (wave & 3) * 32;   // wave M offset within block tile
    const int nw = (wave >> 2) * 64;  // wave N offset within block tile
    const int m0 = blockIdx.y * GBM;
    const int n0 = blockIdx.x * GBN;

    __shared__ __align__(16) unsigned short Abuf[2][GBM][ALD];
    __shared__ __align__(16) unsigned short Bbuf[2][GBK][BLD];

    // staging: each thread moves one 32B chunk of A and one 32B chunk of B
    const int arow = tid >> 1;        // 0..127
    const int acol = (tid & 1) * 16;  // 0 or 16
    const int brow = tid >> 3;        // 0..31
    const int bcol = (tid & 7) * 16;  // 0..112

    const unsigned short* gA = A + (size_t)(m0 + arow) * K + acol;
    const unsigned short* gB = B + (size_t)brow * N + n0 + bcol;

    auto stage = [&](int buf, int kb) {
        const unsigned short* sa = gA + kb;
        const unsigned short* sb = gB + (size_t)kb * N;
        unsigned short* da = &Abuf[buf][arow][acol];
        unsigned short* db = &Bbuf[buf][brow][bcol];
#ifdef USE_ASYNC_LDS
        __builtin_amdgcn_global_load_async_to_lds_b128(to_gbl(sa), to_lds(da), 0, 0);
        __builtin_amdgcn_global_load_async_to_lds_b128(to_gbl(sa), to_lds(da), 16, 0);
        __builtin_amdgcn_global_load_async_to_lds_b128(to_gbl(sb), to_lds(db), 0, 0);
        __builtin_amdgcn_global_load_async_to_lds_b128(to_gbl(sb), to_lds(db), 16, 0);
#else
        *(v16us*)da = *(const v16us*)sa;
        *(v16us*)db = *(const v16us*)sb;
#endif
    };

    v8f c[2][4];
#pragma unroll
    for (int s = 0; s < 2; ++s)
#pragma unroll
        for (int i = 0; i < 4; ++i) c[s][i] = (v8f){};

    stage(0, 0);
    int buf = 0;
    for (int kb = 0; kb < K; kb += GBK) {
#ifdef USE_ASYNC_LDS
        __builtin_amdgcn_s_wait_asynccnt(0);
#endif
        __syncthreads();   // staged data visible; previous buffer free to reuse
        if (kb + GBK < K) stage(buf ^ 1, kb + GBK);

        // A fragments (2 x 16 rows), reused across 4 B fragments each
        v16bf af[2];
#pragma unroll
        for (int s = 0; s < 2; ++s) {
            const unsigned short* ap = &Abuf[buf][mw + s * 16 + l15][half * 8];
            v8us lo = *(const v8us*)(ap);
            v8us hi = *(const v8us*)(ap + 16);
            af[s] = as_bf16(cat8(lo, hi));
        }
#pragma unroll
        for (int i = 0; i < 4; ++i) {
            const unsigned short* bp = &Bbuf[buf][half * 8][nw + i * 16 + l15];
            v16us bu;
#pragma unroll
            for (int e = 0; e < 16; ++e) bu[e] = bp[(e < 8 ? e : e + 8) * BLD];
            v16bf bf = as_bf16(bu);
#pragma unroll
            for (int s = 0; s < 2; ++s)
                c[s][i] = wmma_bf16(af[s], bf, c[s][i]);
        }
        __syncthreads();   // done reading buf before it is restaged next iter
        buf ^= 1;
    }

#pragma unroll
    for (int s = 0; s < 2; ++s) {
        const int mrow = m0 + mw + s * 16 + (half << 3);
#pragma unroll
        for (int i = 0; i < 4; ++i) {
            const int n = n0 + nw + i * 16 + l15;
            const float bv = bias ? bias[n] : 0.0f;
#pragma unroll
            for (int r = 0; r < 8; ++r)
                C[(size_t)(mrow + r) * N + n] = c[s][i][r] + bv;
        }
    }
}

// ---------------------------------------------------------------------------
// Row LayerNorm over 1024 features; fp32 in -> bf16 out.
// ---------------------------------------------------------------------------
__global__ __launch_bounds__(256) void ln_bf16_kernel(
    const float* __restrict__ Z, const float* __restrict__ scale,
    const float* __restrict__ bias, unsigned short* __restrict__ Y) {
    const int row = blockIdx.x;
    const int lane = threadIdx.x & 31;
    const int wave = threadIdx.x >> 5;
    __shared__ float sA[8], sB[8], sStat[2];

    const float* z = Z + (size_t)row * DM;
    float v[4], s = 0.0f, ss = 0.0f;
#pragma unroll
    for (int i = 0; i < 4; ++i) {
        v[i] = z[threadIdx.x + 256 * i];
        s += v[i];
        ss += v[i] * v[i];
    }
#pragma unroll
    for (int off = 16; off > 0; off >>= 1) {
        s  += __shfl_xor(s,  off, 32);
        ss += __shfl_xor(ss, off, 32);
    }
    if (lane == 0) { sA[wave] = s; sB[wave] = ss; }
    __syncthreads();
    if (threadIdx.x == 0) {
        float a = 0.0f, b = 0.0f;
#pragma unroll
        for (int i = 0; i < 8; ++i) { a += sA[i]; b += sB[i]; }
        float mu = a * (1.0f / DM);
        float var = b * (1.0f / DM) - mu * mu;
        sStat[0] = mu;
        sStat[1] = rsqrtf(var + 1e-6f);
    }
    __syncthreads();
    const float mu = sStat[0], rs = sStat[1];
#pragma unroll
    for (int i = 0; i < 4; ++i) {
        int col = threadIdx.x + 256 * i;
        float y = (v[i] - mu) * rs * scale[col] + bias[col];
        Y[(size_t)row * DM + col] = f2bf(y);
    }
}

// ---------------------------------------------------------------------------
// Flash attention: one wave per (b, h, 16 q-rows). Key tiles of 32.
//   S^T = K * Q^T  (A = K rows, coalesced; B = Q^T, hoisted)
//   The S^T C-fragment layout (lane = q-col, keys = VGPR rows split by lane
//   half) is EXACTLY the A-fragment layout of P for P*V -> in-register
//   transpose, no LDS round-trip.
// ---------------------------------------------------------------------------
__global__ __launch_bounds__(256) void attn_kernel(
    const unsigned short* __restrict__ Q, const unsigned short* __restrict__ Kh,
    const unsigned short* __restrict__ V, const int* __restrict__ mask,
    unsigned short* __restrict__ O) {
    const int lane = threadIdx.x & 31;
    const int wave = threadIdx.x >> 5;
    const int half = lane >> 4;
    const int l15  = lane & 15;
    const int bh = blockIdx.x;
    const int b  = bh >> 4;
    const int h  = bh & (HEADS - 1);
    const int q0 = (blockIdx.y * 8 + wave) * 16;

    // Hoisted Q^T B-fragments: two k-steps over d (0..31, 32..63)
    v16bf bq[2];
#pragma unroll
    for (int t = 0; t < 2; ++t) {
        int d = t * 32 + half * 8;
        const unsigned short* qp =
            Q + ((size_t)(b * SEQ + q0 + l15)) * DM + h * DK + d;
        v8us lo = *(const v8us*)(qp);
        v8us hi = *(const v8us*)(qp + 16);
        bq[t] = as_bf16(cat8(lo, hi));
    }

    v8f acc[4];
#pragma unroll
    for (int i = 0; i < 4; ++i) acc[i] = (v8f){};
    float mrun = -1.0e30f;   // running max for q = q0 + l15
    float lrun = 0.0f;       // per-lane-half partial sum
    const float scl = 0.125f;        // 1/sqrt(64)
    const int qIdx = q0 + l15;

    for (int kt = 0; kt < SEQ; kt += 32) {
        // ---- scores S^T for 32 keys (two 16-key fragments) ----
        v8f s[2];
#pragma unroll
        for (int f = 0; f < 2; ++f) {
            const int keyRow = kt + f * 16 + l15;
            const unsigned short* kp =
                Kh + ((size_t)(b * SEQ + keyRow)) * DM + h * DK;
            v8f sf = {};
#pragma unroll
            for (int t = 0; t < 2; ++t) {
                const unsigned short* p = kp + t * 32 + half * 8;
                v8us lo = *(const v8us*)(p);
                v8us hi = *(const v8us*)(p + 16);
                sf = wmma_bf16(as_bf16(cat8(lo, hi)), bq[t], sf);
            }
            s[f] = sf;
        }

        // ---- mask + scale + tile max ----
        float tmax = -1.0e30f;
#pragma unroll
        for (int f = 0; f < 2; ++f) {
            const int kbase = kt + f * 16 + (half << 3);
            const int* mp = mask + ((size_t)(b * SEQ + qIdx)) * SEQ + kbase;
            int4 ma = *(const int4*)(mp);
            int4 mb = *(const int4*)(mp + 4);
            int mm[8] = {ma.x, ma.y, ma.z, ma.w, mb.x, mb.y, mb.z, mb.w};
#pragma unroll
            for (int r = 0; r < 8; ++r) {
                float sv = s[f][r] * scl;
                sv = (mm[r] == 0) ? -1.0e9f : sv;
                s[f][r] = sv;
                tmax = fmaxf(tmax, sv);
            }
        }
        tmax = fmaxf(tmax, __shfl_xor(tmax, 16, 32));
        const float mnew  = fmaxf(mrun, tmax);
        const float alpha = __expf(mrun - mnew);
        mrun = mnew;
        lrun *= alpha;

        // ---- exp -> P A-fragment (in-register transpose) ----
        v16us up;
#pragma unroll
        for (int f = 0; f < 2; ++f) {
            float psum = 0.0f;
#pragma unroll
            for (int r = 0; r < 8; ++r) {
                float p = __expf(s[f][r] - mnew);
                psum += p;
                up[f * 8 + r] = f2bf(p);
            }
            lrun += psum;
        }
        v16bf ap = as_bf16(up);

        // ---- rescale accumulators: alpha per q-row via shfl broadcast ----
        float arow[8];
#pragma unroll
        for (int r = 0; r < 8; ++r)
            arow[r] = __shfl(alpha, r + (half << 3), 32);
#pragma unroll
        for (int i = 0; i < 4; ++i)
#pragma unroll
            for (int r = 0; r < 8; ++r) acc[i][r] *= arow[r];

        // ---- acc += P(16x32) * V(32x64)  (4 B-fragments over d) ----
        const int krow = kt + (half << 3);
        const unsigned short* vp =
            V + ((size_t)(b * SEQ + krow)) * DM + h * DK;
#pragma unroll
        for (int i = 0; i < 4; ++i) {
            v16us vu;
#pragma unroll
            for (int e = 0; e < 16; ++e)
                vu[e] = vp[(size_t)(e < 8 ? e : e + 8) * DM + i * 16 + l15];
            acc[i] = wmma_bf16(ap, as_bf16(vu), acc[i]);
        }
    }

    // ---- normalize and write bf16 head output ----
    lrun += __shfl_xor(lrun, 16, 32);
    const float linv = 1.0f / lrun;
    float lr[8];
#pragma unroll
    for (int r = 0; r < 8; ++r)
        lr[r] = __shfl(linv, r + (half << 3), 32);
#pragma unroll
    for (int i = 0; i < 4; ++i) {
#pragma unroll
        for (int r = 0; r < 8; ++r) {
            const int qrow = q0 + r + (half << 3);
            O[((size_t)(b * SEQ + qrow)) * DM + h * DK + i * 16 + l15] =
                f2bf(acc[i][r] * lr[r]);
        }
    }
}

// ---------------------------------------------------------------------------
extern "C" void kernel_launch(void* const* d_in, const int* in_sizes, int n_in,
                              void* d_out, int out_size, void* d_ws,
                              size_t ws_size, hipStream_t stream) {
    const float* kv   = (const float*)d_in[0];
    const float* q    = (const float*)d_in[1];
    const int*   mask = (const int*)d_in[2];
    const float* Wq = (const float*)d_in[3];
    const float* bq = (const float*)d_in[4];
    const float* Wk = (const float*)d_in[5];
    const float* bk = (const float*)d_in[6];
    const float* Wv = (const float*)d_in[7];
    const float* bv = (const float*)d_in[8];
    const float* qn_s = (const float*)d_in[9];
    const float* qn_b = (const float*)d_in[10];
    const float* kn_s = (const float*)d_in[11];
    const float* kn_b = (const float*)d_in[12];
    const float* vn_s = (const float*)d_in[13];
    const float* vn_b = (const float*)d_in[14];
    const float* Wo = (const float*)d_in[15];
    const float* bo = (const float*)d_in[16];
    float* out = (float*)d_out;

    // workspace layout (~136 MB)
    char* w = (char*)d_ws;
    const size_t act_bf = (size_t)ROWS * DM * sizeof(unsigned short);
    const size_t wgt_bf = (size_t)DM * DM * sizeof(unsigned short);
    unsigned short* q_bf  = (unsigned short*)w; w += act_bf;
    unsigned short* kv_bf = (unsigned short*)w; w += act_bf;
    unsigned short* wq_bf = (unsigned short*)w; w += wgt_bf;
    unsigned short* wk_bf = (unsigned short*)w; w += wgt_bf;
    unsigned short* wv_bf = (unsigned short*)w; w += wgt_bf;
    unsigned short* wo_bf = (unsigned short*)w; w += wgt_bf;
    float* Z = (float*)w;                       w += (size_t)ROWS * DM * 4;
    unsigned short* Qln = (unsigned short*)w;   w += act_bf;
    unsigned short* Kln = (unsigned short*)w;   w += act_bf;
    unsigned short* Vln = (unsigned short*)w;   w += act_bf;
    unsigned short* Abf = (unsigned short*)w;   w += act_bf;

    const int NE = ROWS * DM;  // 8388608
    const int NW = DM * DM;    // 1048576
    cvt_bf16_kernel<<<NE / 256, 256, 0, stream>>>(q,  q_bf,  NE);
    cvt_bf16_kernel<<<NE / 256, 256, 0, stream>>>(kv, kv_bf, NE);
    cvt_bf16_kernel<<<NW / 256, 256, 0, stream>>>(Wq, wq_bf, NW);
    cvt_bf16_kernel<<<NW / 256, 256, 0, stream>>>(Wk, wk_bf, NW);
    cvt_bf16_kernel<<<NW / 256, 256, 0, stream>>>(Wv, wv_bf, NW);
    cvt_bf16_kernel<<<NW / 256, 256, 0, stream>>>(Wo, wo_bf, NW);

    dim3 gg(DM / GBN, ROWS / GBM);  // (8, 64), 256 threads (8 waves)
    // Q projection + LN
    gemm_bf16_kernel<<<gg, 256, 0, stream>>>(q_bf, wq_bf, bq, Z, ROWS, DM, DM);
    ln_bf16_kernel<<<ROWS, 256, 0, stream>>>(Z, qn_s, qn_b, Qln);
    // K projection + LN
    gemm_bf16_kernel<<<gg, 256, 0, stream>>>(kv_bf, wk_bf, bk, Z, ROWS, DM, DM);
    ln_bf16_kernel<<<ROWS, 256, 0, stream>>>(Z, kn_s, kn_b, Kln);
    // V projection + LN
    gemm_bf16_kernel<<<gg, 256, 0, stream>>>(kv_bf, wv_bf, bv, Z, ROWS, DM, DM);
    ln_bf16_kernel<<<ROWS, 256, 0, stream>>>(Z, vn_s, vn_b, Vln);

    // attention: grid (b*h, q-tiles/8), 8 waves/block, 16 q-rows per wave
    dim3 ga(BS * HEADS, SEQ / (16 * 8));
    attn_kernel<<<ga, 256, 0, stream>>>(Qln, Kln, Vln, mask, Abf);

    // output projection -> fp32 d_out
    gemm_bf16_kernel<<<gg, 256, 0, stream>>>(Abf, wo_bf, bo, out, ROWS, DM, DM);
}
